// GQAAttention_22634477650075
// MI455X (gfx1250) — matrix operand order repaired and checked by reading, not compiled, not run on hardware
//
#include <hip/hip_runtime.h>
#include <hip/hip_bf16.h>

// ---------------- problem constants ----------------
#define S_LEN   2048
#define D_MODEL 4096
#define NQH     32
#define NKVH    8
#define HEADD   128
#define NBATCH  4
#define BS_ROWS (NBATCH * S_LEN)   // 8192 rows of the flattened (B*S, D) activation

typedef __attribute__((ext_vector_type(16))) __bf16 v16bf;
typedef __attribute__((ext_vector_type(8)))  __bf16 v8bf;
typedef __attribute__((ext_vector_type(8)))  float  v8f;

// ---------------- WMMA helpers ----------------
__device__ __forceinline__ v8f wmma_bf16(v16bf a, v16bf b, v8f c) {
  // D(f32 16x16) = A(bf16 16x32) * B(bf16 32x16) + C
  return __builtin_amdgcn_wmma_f32_16x16x32_bf16(false, a, false, b, (short)0, c,
                                                 false, false);
}

// A-fragment: lane<16 holds row (lane&15), K = {kb..kb+7, 16+kb..16+kb+7}, kb = 8*(lane>=16)
__device__ __forceinline__ v16bf frag_a(const __bf16* rowPtr, int kb) {
  union { v16bf v; v8bf h[2]; } u;
  u.h[0] = *(const v8bf*)(rowPtr + kb);
  u.h[1] = *(const v8bf*)(rowPtr + kb + 16);
  return u.v;
}
// B-fragment: lane<16 holds column (lane&15) of B (= row of the NxK matrix),
// K = {8*half..8*half+7, 16+8*half..16+8*half+7} (same order as frag_a)
__device__ __forceinline__ v16bf frag_b(const __bf16* rowPtr, int half) {
  union { v16bf v; v8bf h[2]; } u;
  u.h[0] = *(const v8bf*)(rowPtr + half * 8);
  u.h[1] = *(const v8bf*)(rowPtr + half * 8 + 16);
  return u.v;
}

// ---------------- CDNA5 async global->LDS (ASYNCcnt-tracked, bypasses VGPRs) ----------------
// ISA 15.18.3: GLOBAL_LOAD_ASYNC_TO_LDS_B128 (GV mode: 64-bit addr in VGPR pair,
// VDST = LDS byte offset VGPR). Generic LDS pointers carry the LDS offset in the
// low 32 bits (ISA 10.2).
__device__ __forceinline__ void async_ld_b128(__bf16* lptr, const __bf16* gptr) {
  unsigned loff = (unsigned)(unsigned long long)lptr;
  unsigned long long ga = (unsigned long long)gptr;
  asm volatile("global_load_async_to_lds_b128 %0, %1, off"
               :: "v"(loff), "v"(ga) : "memory");
}
__device__ __forceinline__ void wait_async0() {
  asm volatile("s_wait_asynccnt 0x0" ::: "memory");
}

// ---------------- fp32 -> bf16 conversion ----------------
__global__ __launch_bounds__(256) void cvt_f32_bf16(const float* __restrict__ src,
                                                    __bf16* __restrict__ dst, int n) {
  int i = (blockIdx.x * 256 + threadIdx.x) * 4;
  if (i >= n) return;
  float4 v = *(const float4*)(src + i);
  dst[i + 0] = (__bf16)v.x;
  dst[i + 1] = (__bf16)v.y;
  dst[i + 2] = (__bf16)v.z;
  dst[i + 3] = (__bf16)v.w;
}

// ---------------- RoPE (in-place on bf16 Q or K) ----------------
__global__ __launch_bounds__(256) void rope_bf16(__bf16* __restrict__ X,
                                                 const float* __restrict__ cosT,
                                                 const float* __restrict__ sinT,
                                                 int heads, int total) {
  int i = blockIdx.x * 256 + threadIdx.x;
  if (i >= total) return;
  int pair = i & 63;            // pair index within head (HEAD/2 = 64)
  int t    = i >> 6;
  int h    = t % heads;
  int row  = t / heads;         // b*S + s
  int s    = row & (S_LEN - 1);
  float c  = cosT[s * 64 + pair];
  float sn = sinT[s * 64 + pair];
  size_t base = (((size_t)row * heads + h) * HEADD) + pair * 2;
  float xr = (float)X[base];
  float xi = (float)X[base + 1];
  X[base]     = (__bf16)(xr * c - xi * sn);
  X[base + 1] = (__bf16)(xr * sn + xi * c);
}

// ---------------- bf16 GEMM: C[M x N] = A[M x K] * W[N x K]^T + bias ----------------
// block tile 128x64, K-step 64; double-buffered LDS, async staging,
// 8 waves -> 4x2 wave grid of 32x32 wave tiles, 8 WMMAs per barrier.
#define GM 128
#define GN 64
#define GK 64
#define LDT 72   // padded LDS row stride (halfwords); 72*2 = 144B, 16B aligned

__global__ __launch_bounds__(256) void gemm_bf16(const __bf16* __restrict__ A,
                                                 const __bf16* __restrict__ W,
                                                 const float* __restrict__ bias,
                                                 __bf16* __restrict__ outBF,
                                                 float* __restrict__ outF,
                                                 int M, int N, int Kd) {
  __shared__ __bf16 As[2][GM * LDT];
  __shared__ __bf16 Ws[2][GN * LDT];

  int tid  = threadIdx.x;
  int bm   = blockIdx.y, bn = blockIdx.x;
  int wave = tid >> 5, lane = tid & 31;
  int half = lane >> 4, lr = lane & 15, kb = half * 8;
  int wm = (wave & 3) * 32;       // wave M offset in block tile
  int wn = (wave >> 2) * 32;      // wave N offset in block tile

  const v8f vzero = {0.f, 0.f, 0.f, 0.f, 0.f, 0.f, 0.f, 0.f};
  v8f acc[2][2] = {{vzero, vzero}, {vzero, vzero}};

  // staging assignment: A 128x64 hw -> 32 hw/thread; W 64x64 hw -> 16 hw/thread
  int arow = tid >> 1, acol = (tid & 1) * 32;
  int wrow = tid >> 2, wcol = (tid & 3) * 16;
  const __bf16* Ag = A + (size_t)(bm * GM + arow) * Kd + acol;
  const __bf16* Wg = W + (size_t)(bn * GN + wrow) * Kd + wcol;

  // prologue: stage tile 0 into buffer 0
#pragma unroll
  for (int u = 0; u < 4; ++u)
    async_ld_b128(&As[0][arow * LDT + acol + u * 8], Ag + u * 8);
#pragma unroll
  for (int u = 0; u < 2; ++u)
    async_ld_b128(&Ws[0][wrow * LDT + wcol + u * 8], Wg + u * 8);
  Ag += GK;
  Wg += GK;
  wait_async0();
  __syncthreads();

  int nIter = Kd / GK;
  for (int it = 0; it < nIter; ++it) {
    int cur = it & 1, nxt = cur ^ 1;
    if (it + 1 < nIter) {  // stage next tile asynchronously while computing
#pragma unroll
      for (int u = 0; u < 4; ++u)
        async_ld_b128(&As[nxt][arow * LDT + acol + u * 8], Ag + u * 8);
#pragma unroll
      for (int u = 0; u < 2; ++u)
        async_ld_b128(&Ws[nxt][wrow * LDT + wcol + u * 8], Wg + u * 8);
      Ag += GK;
      Wg += GK;
    }

#pragma unroll
    for (int kk = 0; kk < GK; kk += 32) {
      v16bf af[2], bf_[2];
#pragma unroll
      for (int s = 0; s < 2; ++s) {
        af[s]  = frag_a(&As[cur][(wm + s * 16 + lr) * LDT + kk], kb);
        bf_[s] = frag_b(&Ws[cur][(wn + s * 16 + lr) * LDT + kk], half);
      }
#pragma unroll
      for (int i = 0; i < 2; ++i)
#pragma unroll
        for (int j = 0; j < 2; ++j)
          acc[i][j] = wmma_bf16(af[i], bf_[j], acc[i][j]);
    }

    wait_async0();     // next-tile LDS writes landed
    __syncthreads();   // every wave done with 'cur', 'nxt' visible to all
  }

  // epilogue: C layout -> lane<16: (M=r, N=lr); lane>=16: (M=r+8, N=lr)
#pragma unroll
  for (int j = 0; j < 2; ++j) {
    int gcol = bn * GN + wn + j * 16 + lr;
    float bv = bias ? bias[gcol] : 0.0f;
#pragma unroll
    for (int i = 0; i < 2; ++i) {
#pragma unroll
      for (int r = 0; r < 8; ++r) {
        int grow = bm * GM + wm + i * 16 + r + half * 8;
        float v = acc[i][j][r] + bv;
        if (outF) outF[(size_t)grow * N + gcol] = v;
        else      outBF[(size_t)grow * N + gcol] = (__bf16)v;
      }
    }
  }
}

// ---------------- flash attention (causal, GQA) ----------------
// grid: (S/64, NQH, B); block: 128 threads = 4 waves; each wave: 16 q-rows x HEAD 128
#define KLDT 136  // K tile LDS row stride (128 + 8 pad hw); 272B, 16B aligned
#define VLDT 40   // Vt / P LDS row stride

__global__ __launch_bounds__(128) void flash_gqa(const __bf16* __restrict__ Q,
                                                 const __bf16* __restrict__ K,
                                                 const __bf16* __restrict__ V,
                                                 __bf16* __restrict__ O) {
  __shared__ __bf16 Ks[32 * KLDT];        // K tile, row-major (key x head)
  __shared__ __bf16 Vt[HEADD * VLDT];     // V tile transposed (head x key)
  __shared__ __bf16 Ps[4][16 * VLDT];     // per-wave P staging (16 x 32)

  int qb = blockIdx.x;        // 0..31 : 64-row q block
  int h  = blockIdx.y;        // 0..31 : q head
  int b  = blockIdx.z;        // 0..3
  int kvh = h & (NKVH - 1);   // kv head = h % 8
  int tid = threadIdx.x;
  int wave = tid >> 5, lane = tid & 31;
  int half = lane >> 4, lr = lane & 15, kb = half * 8;

  int qrow0 = qb * 64 + wave * 16;

  // load Q fragments (4 x 16x32 covering HEAD=128) straight from global
  v16bf qf[4];
  {
    const __bf16* qp = Q + ((size_t)(b * S_LEN + qrow0 + lr) * D_MODEL) + h * HEADD;
#pragma unroll
    for (int f = 0; f < 4; ++f) qf[f] = frag_a(qp + f * 32, kb);
  }

  const v8f vzero = {0.f, 0.f, 0.f, 0.f, 0.f, 0.f, 0.f, 0.f};
  v8f acc[8];
#pragma unroll
  for (int n = 0; n < 8; ++n) acc[n] = vzero;
  float mrow[8], lrow[8];
#pragma unroll
  for (int r = 0; r < 8; ++r) { mrow[r] = -1e30f; lrow[r] = 0.0f; }

  const float scale = 0.088388347648318447f;  // 1/sqrt(HEAD)
  int ktiles = (qb + 1) * 2;                  // causal: keys up to qb*64+63, 32/tile
  int srow = tid >> 2;                        // 0..31 key row staged by this thread
  int scol = (tid & 3) * 32;                  // head-dim chunk

  for (int t = 0; t < ktiles; ++t) {
    // stage K tile row-major via async global->LDS; V tile transposed via ds stores
    const __bf16* kp = K + ((size_t)(b * S_LEN + t * 32 + srow) * (NKVH * HEADD)) + kvh * HEADD + scol;
    const __bf16* vp = V + ((size_t)(b * S_LEN + t * 32 + srow) * (NKVH * HEADD)) + kvh * HEADD + scol;
#pragma unroll
    for (int u = 0; u < 4; ++u)
      async_ld_b128(&Ks[srow * KLDT + scol + u * 8], kp + u * 8);
#pragma unroll
    for (int u = 0; u < 4; ++u) {
      v8bf vv = *(const v8bf*)(vp + u * 8);
#pragma unroll
      for (int e = 0; e < 8; ++e)
        Vt[(scol + u * 8 + e) * VLDT + srow] = vv[e];
    }
    if (t + 1 < ktiles) __builtin_prefetch(vp + 32 * NKVH * HEADD, 0, 1);
    wait_async0();
    __syncthreads();

    // scores: S(16x32) = Q(16x128) * Ktile(32x128)^T, two 16x16 column tiles
    v8f s0 = vzero, s1 = vzero;
#pragma unroll
    for (int f = 0; f < 4; ++f) {
      v16bf b0 = frag_b(&Ks[lr * KLDT + f * 32], half);         // keys t*32 + 0..15
      v16bf b1 = frag_b(&Ks[(16 + lr) * KLDT + f * 32], half);  // keys t*32 + 16..31
      s0 = wmma_bf16(qf[f], b0, s0);
      s1 = wmma_bf16(qf[f], b1, s1);
    }

    // online softmax + P staging
    __bf16* Pw = Ps[wave];
#pragma unroll
    for (int r = 0; r < 8; ++r) {
      int qi  = qrow0 + r + half * 8;
      int kj0 = t * 32 + lr;
      int kj1 = kj0 + 16;
      float x0 = s0[r] * scale + ((kj0 > qi) ? -1e30f : 0.0f);
      float x1 = s1[r] * scale + ((kj1 > qi) ? -1e30f : 0.0f);
      float mx = fmaxf(x0, x1);
#pragma unroll
      for (int m = 8; m >= 1; m >>= 1) mx = fmaxf(mx, __shfl_xor(mx, m, 32));
      float mnew  = fmaxf(mrow[r], mx);
      float alpha = __expf(mrow[r] - mnew);
      float p0 = __expf(x0 - mnew);
      float p1 = __expf(x1 - mnew);
      float rs = p0 + p1;
#pragma unroll
      for (int m = 8; m >= 1; m >>= 1) rs += __shfl_xor(rs, m, 32);
      lrow[r] = lrow[r] * alpha + rs;
      mrow[r] = mnew;
#pragma unroll
      for (int n = 0; n < 8; ++n) acc[n][r] *= alpha;
      Pw[(r + half * 8) * VLDT + lr]      = (__bf16)p0;
      Pw[(r + half * 8) * VLDT + 16 + lr] = (__bf16)p1;
    }
    // per-wave LDS RAW: same-wave DS ops are tracked by the compiler via DScnt

    // O += P(16x32) * Vtile(32x128): one A frag, 8 column tiles
    v16bf pa = frag_a(&Pw[lr * VLDT], kb);
#pragma unroll
    for (int n = 0; n < 8; ++n) {
      v16bf vb = frag_b(&Vt[(n * 16 + lr) * VLDT], half);
      acc[n] = wmma_bf16(pa, vb, acc[n]);
    }
    __syncthreads();
  }

  // normalize and store O (bf16), layout (b*S + row, 4096) head-major
  __bf16* Ob = O + ((size_t)(b * S_LEN + qrow0) * D_MODEL) + h * HEADD;
#pragma unroll
  for (int n = 0; n < 8; ++n) {
#pragma unroll
    for (int r = 0; r < 8; ++r) {
      int M = r + half * 8;
      float v = acc[n][r] / lrow[r];
      Ob[(size_t)M * D_MODEL + n * 16 + lr] = (__bf16)v;
    }
  }
}

// ---------------- host launch ----------------
extern "C" void kernel_launch(void* const* d_in, const int* in_sizes, int n_in,
                              void* d_out, int out_size, void* d_ws, size_t ws_size,
                              hipStream_t stream) {
  const float* input = (const float*)d_in[0];
  const float* fcos  = (const float*)d_in[1];
  const float* fsin  = (const float*)d_in[2];
  const float* qw    = (const float*)d_in[3];
  const float* qb    = (const float*)d_in[4];
  const float* kw    = (const float*)d_in[5];
  const float* kbias = (const float*)d_in[6];
  const float* vw    = (const float*)d_in[7];
  const float* vbias = (const float*)d_in[8];
  const float* ow    = (const float*)d_in[9];
  const float* obias = (const float*)d_in[10];
  float* out = (float*)d_out;

  const size_t nIn  = (size_t)BS_ROWS * D_MODEL;        // 33.5M
  const size_t nQW  = (size_t)D_MODEL * D_MODEL;        // 16.8M
  const size_t nKW  = (size_t)(NKVH * HEADD) * D_MODEL; // 4.2M
  const size_t nKV  = (size_t)BS_ROWS * (NKVH * HEADD); // 8.4M

  char* w = (char*)d_ws;
  __bf16* inBF = (__bf16*)w; w += nIn * 2;
  __bf16* qwBF = (__bf16*)w; w += nQW * 2;
  __bf16* kwBF = (__bf16*)w; w += nKW * 2;
  __bf16* vwBF = (__bf16*)w; w += nKW * 2;
  __bf16* owBF = (__bf16*)w; w += nQW * 2;
  __bf16* Qb   = (__bf16*)w; w += nIn * 2;
  __bf16* Kb   = (__bf16*)w; w += nKV * 2;
  __bf16* Vb   = (__bf16*)w; w += nKV * 2;
  __bf16* Ob   = (__bf16*)w; w += nIn * 2;

  // 1) fp32 -> bf16 conversions
  cvt_f32_bf16<<<(unsigned)(nIn / 4 / 256), 256, 0, stream>>>(input, inBF, (int)nIn);
  cvt_f32_bf16<<<(unsigned)(nQW / 4 / 256), 256, 0, stream>>>(qw, qwBF, (int)nQW);
  cvt_f32_bf16<<<(unsigned)(nKW / 4 / 256), 256, 0, stream>>>(kw, kwBF, (int)nKW);
  cvt_f32_bf16<<<(unsigned)(nKW / 4 / 256), 256, 0, stream>>>(vw, vwBF, (int)nKW);
  cvt_f32_bf16<<<(unsigned)(nQW / 4 / 256), 256, 0, stream>>>(ow, owBF, (int)nQW);

  // 2) projections  X * W^T + b
  gemm_bf16<<<dim3(D_MODEL / GN, BS_ROWS / GM), 256, 0, stream>>>(
      inBF, qwBF, qb, Qb, nullptr, BS_ROWS, D_MODEL, D_MODEL);
  gemm_bf16<<<dim3((NKVH * HEADD) / GN, BS_ROWS / GM), 256, 0, stream>>>(
      inBF, kwBF, kbias, Kb, nullptr, BS_ROWS, NKVH * HEADD, D_MODEL);
  gemm_bf16<<<dim3((NKVH * HEADD) / GN, BS_ROWS / GM), 256, 0, stream>>>(
      inBF, vwBF, vbias, Vb, nullptr, BS_ROWS, NKVH * HEADD, D_MODEL);

  // 3) RoPE on Q and K
  {
    int totQ = BS_ROWS * NQH * (HEADD / 2);
    int totK = BS_ROWS * NKVH * (HEADD / 2);
    rope_bf16<<<(totQ + 255) / 256, 256, 0, stream>>>(Qb, fcos, fsin, NQH, totQ);
    rope_bf16<<<(totK + 255) / 256, 256, 0, stream>>>(Kb, fcos, fsin, NKVH, totK);
  }

  // 4) causal flash attention
  flash_gqa<<<dim3(S_LEN / 64, NQH, NBATCH), 128, 0, stream>>>(Qb, Kb, Vb, Ob);

  // 5) output projection -> fp32 d_out
  gemm_bf16<<<dim3(D_MODEL / GN, BS_ROWS / GM), 256, 0, stream>>>(
      Ob, owBF, obias, nullptr, out, BS_ROWS, D_MODEL, D_MODEL);
}
